// InteractionLSTM_Cell_23175643529763
// MI455X (gfx1250) — matrix operand order repaired and checked by reading, not compiled, not run on hardware
//
#include <hip/hip_runtime.h>
#include <hip/hip_bf16.h>
#include <math.h>

typedef __attribute__((ext_vector_type(16))) _Float16 v16h;
typedef __attribute__((ext_vector_type(8)))  float    v8f;

#define B_   4
#define N_   64
#define D_   64
#define F_   15
#define U_   2056
#define HD_  512
#define BD_  1680
#define XD_  2704
#define M_   (B_*N_)   /* 256 rows in every dense GEMM */

__device__ __forceinline__ float gelu_f(float x) {
    return 0.5f * x * (1.0f + erff(x * 0.70710678118654752f));
}
__device__ __forceinline__ float elu_f(float x) {
    return x > 0.0f ? x : expm1f(x);
}
__device__ __forceinline__ float sigm_f(float x) {
    return 1.0f / (1.0f + expf(-x));
}

// pack 4 floats -> 4 f16 in a uint2 (for ds_store_b64)
__device__ __forceinline__ uint2 pack4h(float a, float b, float c, float d) {
    union { _Float16 h[4]; uint2 u; } p;
    p.h[0] = (_Float16)a; p.h[1] = (_Float16)b;
    p.h[2] = (_Float16)c; p.h[3] = (_Float16)d;
    return p.u;
}

// ---------------------------------------------------------------------------
// Kernel 0: buf = object_buffer reshaped (B,F,N,D), buf[:,t] += inputs
// ---------------------------------------------------------------------------
__global__ void buf_update_kernel(const float* __restrict__ inputs,
                                  const int* __restrict__ tptr,
                                  const float* __restrict__ object_buffer,
                                  float* __restrict__ buf_out)
{
    int q = blockIdx.x * 256 + threadIdx.x;
    if (q >= B_ * F_ * N_ * D_) return;
    int t  = *tptr;
    int b  = q / (F_ * N_ * D_);
    int r  = q % (F_ * N_ * D_);
    int f  = r / (N_ * D_);
    int nd = r % (N_ * D_);
    float v = object_buffer[q];
    if (f == t) v += inputs[b * (N_ * D_) + nd];
    buf_out[q] = v;
}

// ---------------------------------------------------------------------------
// Kernel 1: zb_in[b,n, d*F+f] = buf[b,f,n,d]   (transpose-gather, 256x960)
// ---------------------------------------------------------------------------
__global__ void zbin_kernel(const float* __restrict__ buf,
                            float* __restrict__ zb_in)
{
    int q = blockIdx.x * 256 + threadIdx.x;
    if (q >= M_ * (D_ * F_)) return;
    int rowN = q / (D_ * F_);     // b*64 + n
    int k    = q % (D_ * F_);
    int b = rowN >> 6, n = rowN & 63;
    int d = k / F_,   f = k % F_;
    zb_in[q] = buf[(((size_t)b * F_ + f) * N_ + n) * D_ + d];
}

// ---------------------------------------------------------------------------
// WMMA micro-step shared by the generic GEMM (64x64 block tile, 4x2 waves)
// ---------------------------------------------------------------------------
__device__ __forceinline__ void gemm_mma_step(const _Float16* sA, const _Float16* sB,
                                              int wm, int wn, int g, int nl,
                                              v8f& c0, v8f& c1)
{
    union { v16h v; uint4 q[2]; } af, b0, b1;
    af.q[0] = *(const uint4*)&sA[(wm * 16 + nl) * 32 + g * 16];
    af.q[1] = *(const uint4*)&sA[(wm * 16 + nl) * 32 + g * 16 + 8];
    b0.q[0] = *(const uint4*)&sB[(wn * 32 +      nl) * 32 + g * 16];
    b0.q[1] = *(const uint4*)&sB[(wn * 32 +      nl) * 32 + g * 16 + 8];
    b1.q[0] = *(const uint4*)&sB[(wn * 32 + 16 + nl) * 32 + g * 16];
    b1.q[1] = *(const uint4*)&sB[(wn * 32 + 16 + nl) * 32 + g * 16 + 8];
    c0 = __builtin_amdgcn_wmma_f32_16x16x32_f16(false, af.v, false, b0.v,
                                                (short)0, c0, false, false);
    c1 = __builtin_amdgcn_wmma_f32_16x16x32_f16(false, af.v, false, b1.v,
                                                (short)0, c1, false, false);
}

// ---------------------------------------------------------------------------
// Generic WMMA GEMM:  C[M x N] = act(A[M x K] @ W[K x N] + bias)
//   A row-major (lda), W row-major (ldw == N), C row-major (ldc).
//   M multiple of 64 (gridDim.y*64). Guard-free main K loop + guarded tail;
//   uniform fullN branch so interior blocks are branchless.
//   act: 0 = none, 1 = ELU.
// ---------------------------------------------------------------------------
__global__ __launch_bounds__(256)
void gemm_wmma_f16(const float* __restrict__ A, int lda,
                   const float* __restrict__ W, int N, int K,
                   const float* __restrict__ bias,
                   float* __restrict__ C, int ldc, int act)
{
    __shared__ _Float16 sA[64 * 32];        // [row][k]
    __shared__ _Float16 sB[64 * 32];        // transposed: [n][k]

    const int tid  = threadIdx.x;
    const int wave = tid >> 5, lane = tid & 31;
    const int g    = lane >> 4, nl = lane & 15;
    const int wm   = wave >> 1, wn = wave & 1;   // 4 x 2 wave grid
    const int rowBase = blockIdx.y * 64;
    const int colBase = blockIdx.x * 64;
    const int Kmain   = K & ~31;
    const bool fullN  = (colBase + 64 <= N);

    v8f c0 = {};
    v8f c1 = {};

    for (int k0 = 0; k0 < Kmain; k0 += 32) {
        // stage A tile 64x32: float4 global loads, packed f16 b64 LDS stores
        #pragma unroll
        for (int q = 0; q < 2; ++q) {
            int i = (tid + q * 256) * 4;            // 0..2044, step 4
            int r = i >> 5, kk = i & 31;
            float4 v = *(const float4*)&A[(size_t)(rowBase + r) * lda + k0 + kk];
            *(uint2*)&sA[r * 32 + kk] = pack4h(v.x, v.y, v.z, v.w);
        }
        // stage B tile (64 cols x 32 k) transposed [n][k]
        if (fullN) {
            #pragma unroll
            for (int q = 0; q < 2; ++q) {
                int i = (tid + q * 256) * 4;
                int n = i & 63, kk = i >> 6;
                float4 v = *(const float4*)&W[(size_t)(k0 + kk) * N + colBase + n];
                sB[(n + 0) * 32 + kk] = (_Float16)v.x;
                sB[(n + 1) * 32 + kk] = (_Float16)v.y;
                sB[(n + 2) * 32 + kk] = (_Float16)v.z;
                sB[(n + 3) * 32 + kk] = (_Float16)v.w;
            }
        } else {
            #pragma unroll
            for (int q = 0; q < 8; ++q) {
                int i = tid + q * 256;
                int n = i & 63, kk = i >> 6;
                float v = (colBase + n < N) ? W[(size_t)(k0 + kk) * N + colBase + n]
                                            : 0.0f;
                sB[n * 32 + kk] = (_Float16)v;
            }
        }
        if (k0 + 32 < Kmain)
            __builtin_prefetch(&W[(size_t)(k0 + 32) * N + colBase + (tid & 63)], 0, 0);
        __syncthreads();
        gemm_mma_step(sA, sB, wm, wn, g, nl, c0, c1);
        __syncthreads();
    }

    if (Kmain < K) {            // guarded K tail (one step)
        int k0 = Kmain;
        #pragma unroll
        for (int q = 0; q < 8; ++q) {
            int i = tid + q * 256;
            int r = i >> 5, kk = i & 31;
            float v = (k0 + kk < K) ? A[(size_t)(rowBase + r) * lda + k0 + kk] : 0.0f;
            sA[r * 32 + kk] = (_Float16)v;
        }
        #pragma unroll
        for (int q = 0; q < 8; ++q) {
            int i = tid + q * 256;
            int n = i & 63, kk = i >> 6;
            float v = ((k0 + kk < K) && (colBase + n < N))
                        ? W[(size_t)(k0 + kk) * N + colBase + n] : 0.0f;
            sB[n * 32 + kk] = (_Float16)v;
        }
        __syncthreads();
        gemm_mma_step(sA, sB, wm, wn, g, nl, c0, c1);
        __syncthreads();
    }

    // epilogue
    if (fullN) {
        #pragma unroll
        for (int r = 0; r < 8; ++r) {
            int row  = rowBase + wm * 16 + r + g * 8;
            int col0 = colBase + wn * 32 + nl;
            int col1 = col0 + 16;
            float v0 = c0[r] + (bias ? bias[col0] : 0.0f);
            float v1 = c1[r] + (bias ? bias[col1] : 0.0f);
            if (act == 1) { v0 = elu_f(v0); v1 = elu_f(v1); }
            C[(size_t)row * ldc + col0] = v0;
            C[(size_t)row * ldc + col1] = v1;
        }
    } else {
        #pragma unroll
        for (int r = 0; r < 8; ++r) {
            int row  = rowBase + wm * 16 + r + g * 8;
            int col0 = colBase + wn * 32 + nl;
            int col1 = col0 + 16;
            if (col0 < N) {
                float v = c0[r] + (bias ? bias[col0] : 0.0f);
                if (act == 1) v = elu_f(v);
                C[(size_t)row * ldc + col0] = v;
            }
            if (col1 < N) {
                float v = c1[r] + (bias ? bias[col1] : 0.0f);
                if (act == 1) v = elu_f(v);
                C[(size_t)row * ldc + col1] = v;
            }
        }
    }
}

// ---------------------------------------------------------------------------
// Fused pairwise interaction MLP + sum/max reduction over j.
// One block per (b,i); j in 16-row WMMA tiles; each wave owns 32 columns.
// ---------------------------------------------------------------------------
__device__ __forceinline__ void pair_layer(
    const _Float16* sA,                 // 16 x 512 f16 row-major
    _Float16* sB,                       // 256 x 32 staging, [n][k]
    const float* __restrict__ W,        // 512 x 512
    const float* __restrict__ bias,     // 512
    _Float16* outF16,                   // if non-null: gelu -> f16 (16x512)
    float* outF32, int addF32,          // else gelu -> f32 write/add
    int tid)
{
    const int wave = tid >> 5, lane = tid & 31;
    const int g = lane >> 4, nl = lane & 15;

    for (int p = 0; p < 2; ++p) {           // 2 passes of 256 columns
        v8f a0 = {};
        v8f a1 = {};
        for (int ks = 0; ks < 16; ++ks) {   // K = 512 in steps of 32
            int k0 = ks * 32;
            // stage W[k0:k0+32, p*256:+256] transposed, float4 loads
            #pragma unroll
            for (int q = 0; q < 8; ++q) {
                int i = (tid + q * 256) * 4;      // 0..8188, step 4
                int n = i & 255, kk = i >> 8;
                float4 v = *(const float4*)&W[(size_t)(k0 + kk) * HD_ + p * 256 + n];
                sB[(n + 0) * 32 + kk] = (_Float16)v.x;
                sB[(n + 1) * 32 + kk] = (_Float16)v.y;
                sB[(n + 2) * 32 + kk] = (_Float16)v.z;
                sB[(n + 3) * 32 + kk] = (_Float16)v.w;
            }
            if (ks < 15)
                __builtin_prefetch(&W[(size_t)(k0 + 32) * HD_ + p * 256 + (tid & 255)], 0, 0);
            __syncthreads();
            union { v16h v; uint4 q[2]; } af, b0, b1;
            af.q[0] = *(const uint4*)&sA[nl * HD_ + k0 + g * 16];
            af.q[1] = *(const uint4*)&sA[nl * HD_ + k0 + g * 16 + 8];
            b0.q[0] = *(const uint4*)&sB[(wave * 32 +      nl) * 32 + g * 16];
            b0.q[1] = *(const uint4*)&sB[(wave * 32 +      nl) * 32 + g * 16 + 8];
            b1.q[0] = *(const uint4*)&sB[(wave * 32 + 16 + nl) * 32 + g * 16];
            b1.q[1] = *(const uint4*)&sB[(wave * 32 + 16 + nl) * 32 + g * 16 + 8];
            a0 = __builtin_amdgcn_wmma_f32_16x16x32_f16(false, af.v, false, b0.v,
                                                        (short)0, a0, false, false);
            a1 = __builtin_amdgcn_wmma_f32_16x16x32_f16(false, af.v, false, b1.v,
                                                        (short)0, a1, false, false);
            __syncthreads();
        }
        #pragma unroll
        for (int r = 0; r < 8; ++r) {
            int row  = r + g * 8;
            int col0 = p * 256 + wave * 32 + nl;
            int col1 = col0 + 16;
            float v0 = gelu_f(a0[r] + bias[col0]);
            float v1 = gelu_f(a1[r] + bias[col1]);
            if (outF16) {
                outF16[row * HD_ + col0] = (_Float16)v0;
                outF16[row * HD_ + col1] = (_Float16)v1;
            } else if (addF32) {
                outF32[row * HD_ + col0] += v0;
                outF32[row * HD_ + col1] += v1;
            } else {
                outF32[row * HD_ + col0] = v0;
                outF32[row * HD_ + col1] = v1;
            }
        }
        __syncthreads();
    }
}

// build gelu(base_row + other_j + bias) -> 16x512 f16 tile, vectorized
__device__ __forceinline__ void build_layer1(
    _Float16* sA1, const float* sBase,          // sBase: 512 f32 in LDS
    const float* __restrict__ gj,               // [256 x 512] global
    const float* __restrict__ bias,             // 512
    int b, int jt, int tid)
{
    #pragma unroll
    for (int q = 0; q < 8; ++q) {
        int idx = (tid + q * 256) * 4;          // 0..8188, step 4
        int jj = idx >> 9, c = idx & 511;
        int j = jt * 16 + jj;
        float4 gv = *(const float4*)&gj[(size_t)(b * 64 + j) * HD_ + c];
        float4 bv = *(const float4*)&bias[c];
        float4 rv = *(const float4*)&sBase[c];
        *(uint2*)&sA1[idx] = pack4h(gelu_f(rv.x + gv.x + bv.x),
                                    gelu_f(rv.y + gv.y + bv.y),
                                    gelu_f(rv.z + gv.z + bv.z),
                                    gelu_f(rv.w + gv.w + bv.w));
    }
}

__global__ __launch_bounds__(256)
void pairwise_kernel(const float* __restrict__ r_i, const float* __restrict__ r_j,
                     const float* __restrict__ l_i, const float* __restrict__ l_j,
                     const float* __restrict__ br1, const float* __restrict__ Wr2,
                     const float* __restrict__ br2, const float* __restrict__ Wr3,
                     const float* __restrict__ br3, const float* __restrict__ bl1,
                     const float* __restrict__ Wl2, const float* __restrict__ bl2,
                     const float* __restrict__ Wl3, const float* __restrict__ bl3,
                     float* __restrict__ X)
{
    extern __shared__ __align__(16) char smem[];     // 90112 bytes dynamic LDS
    _Float16* sA1    = (_Float16*)smem;              // 16*512 f16 = 16KB
    _Float16* sT2    = sA1 + 16 * HD_;               // 16*512 f16 = 16KB
    _Float16* sB     = sT2 + 16 * HD_;               // 256*32 f16 = 16KB
    float*    sInter = (float*)(sB + 256 * 32);      // 16*512 f32 = 32KB
    float*    sSum   = sInter + 16 * HD_;            // 512 f32
    float*    sMax   = sSum + HD_;                   // 512 f32
    float*    sRi    = sMax + HD_;                   // 512 f32
    float*    sLi    = sRi + HD_;                    // 512 f32

    const int tid = threadIdx.x;
    const int row = blockIdx.x;       // b*64 + i
    const int b   = row >> 6;

    for (int c = tid; c < HD_; c += 256) {
        sRi[c]  = r_i[(size_t)row * HD_ + c];
        sLi[c]  = l_i[(size_t)row * HD_ + c];
        sSum[c] = 0.0f;
        sMax[c] = -INFINITY;
    }
    __syncthreads();

    for (int jt = 0; jt < 4; ++jt) {
        // rho: gelu(r_i + r_j + br1) -> 2-layer MLP -> sInter (write)
        build_layer1(sA1, sRi, r_j, br1, b, jt, tid);
        __syncthreads();
        pair_layer(sA1, sB, Wr2, br2, sT2, nullptr, 0, tid);
        __syncthreads();
        pair_layer(sT2, sB, Wr3, br3, nullptr, sInter, 0, tid);
        __syncthreads();

        // lambda: gelu(l_i + l_j + bl1) -> 2-layer MLP -> sInter (add)
        build_layer1(sA1, sLi, l_j, bl1, b, jt, tid);
        __syncthreads();
        pair_layer(sA1, sB, Wl2, bl2, sT2, nullptr, 0, tid);
        __syncthreads();
        pair_layer(sT2, sB, Wl3, bl3, nullptr, sInter, 1, tid);
        __syncthreads();

        // reduce the 16 j-rows into running sum/max
        for (int c = tid; c < HD_; c += 256) {
            float s = 0.0f, mx = -INFINITY;
            #pragma unroll
            for (int jj = 0; jj < 16; ++jj) {
                float v = sInter[jj * HD_ + c];
                s += v;
                mx = fmaxf(mx, v);
            }
            sSum[c] += s;
            sMax[c]  = fmaxf(sMax[c], mx);
        }
        __syncthreads();
    }

    for (int c = tid; c < HD_; c += 256) {
        X[(size_t)row * XD_ + BD_ + c]       = sSum[c];
        X[(size_t)row * XD_ + BD_ + HD_ + c] = sMax[c];
    }
}

// ---------------------------------------------------------------------------
// LSTM elementwise combine (faithful: gate = sig(XW) + sig(HW))
// ---------------------------------------------------------------------------
__global__ void lstm_kernel(const float* __restrict__ gxi, const float* __restrict__ gxf,
                            const float* __restrict__ gxo, const float* __restrict__ gxc,
                            const float* __restrict__ ghi, const float* __restrict__ ghf,
                            const float* __restrict__ gho, const float* __restrict__ ghc,
                            const float* __restrict__ cells,
                            float* __restrict__ Hout, float* __restrict__ Cout)
{
    int q = blockIdx.x * 256 + threadIdx.x;
    if (q >= M_ * U_) return;
    float I  = sigm_f(gxi[q]) + sigm_f(ghi[q]);
    float Fg = sigm_f(gxf[q]) + sigm_f(ghf[q]);
    float O  = sigm_f(gxo[q]) + sigm_f(gho[q]);
    float Ct = tanhf(gxc[q]) + tanhf(ghc[q]);
    float Cn = Fg * cells[q] + I * Ct;
    Hout[q] = O * tanhf(Cn);
    Cout[q] = Cn;
}

// ---------------------------------------------------------------------------
// Host-side orchestration
// ---------------------------------------------------------------------------
extern "C" void kernel_launch(void* const* d_in, const int* in_sizes, int n_in,
                              void* d_out, int out_size, void* d_ws, size_t ws_size,
                              hipStream_t stream)
{
    (void)in_sizes; (void)n_in; (void)out_size; (void)ws_size;

    const float* inputs = (const float*)d_in[0];
    const int*   tptr   = (const int*)  d_in[1];
    const float* objbuf = (const float*)d_in[2];
    const float* cells  = (const float*)d_in[3];
    const float* hidden = (const float*)d_in[4];
    // params in dict order
    const float* Wb   = (const float*)d_in[5];
    const float* bb   = (const float*)d_in[6];
    const float* Wr1a = (const float*)d_in[7];
    const float* Wr1b = (const float*)d_in[8];
    const float* br1  = (const float*)d_in[9];
    const float* Wr2  = (const float*)d_in[10];
    const float* br2  = (const float*)d_in[11];
    const float* Wr3  = (const float*)d_in[12];
    const float* br3  = (const float*)d_in[13];
    const float* Wl1a = (const float*)d_in[14];
    const float* Wl1b = (const float*)d_in[15];
    const float* bl1  = (const float*)d_in[16];
    const float* Wl2  = (const float*)d_in[17];
    const float* bl2  = (const float*)d_in[18];
    const float* Wl3  = (const float*)d_in[19];
    const float* bl3  = (const float*)d_in[20];
    const float* Wxi  = (const float*)d_in[21];
    const float* bxi  = (const float*)d_in[22];
    const float* Wxf  = (const float*)d_in[23];
    const float* bxf  = (const float*)d_in[24];
    const float* Wxo  = (const float*)d_in[25];
    const float* bxo  = (const float*)d_in[26];
    const float* Wxc  = (const float*)d_in[27];
    const float* bxc  = (const float*)d_in[28];
    const float* Whi  = (const float*)d_in[29];
    const float* bhi  = (const float*)d_in[30];
    const float* Whf  = (const float*)d_in[31];
    const float* bhf  = (const float*)d_in[32];
    const float* Who  = (const float*)d_in[33];
    const float* bho  = (const float*)d_in[34];
    const float* Whc  = (const float*)d_in[35];
    const float* bhc  = (const float*)d_in[36];
    const float* Wd   = (const float*)d_in[37];
    const float* bd   = (const float*)d_in[38];

    // d_out layout: [output 16384 | buf 245760 | H_new 526336 | C_new 526336]
    float* out    = (float*)d_out;
    float* outDec = out;
    float* bufOut = out + 16384;
    float* Hout   = out + 262144;
    float* Cout   = out + 788480;

    // workspace layout (floats)
    float* ws    = (float*)d_ws;
    float* zb_in = ws;                    // 256*960   = 245760
    float* X     = zb_in + 245760;        // 256*2704  = 692224
    float* r_i   = X + 692224;            // 256*512
    float* r_j   = r_i + 131072;
    float* l_i   = r_j + 131072;
    float* l_j   = l_i + 131072;
    float* gx    = l_j + 131072;          // 4 * 256*2056
    float* gh    = gx + 4 * 526336;       // 4 * 256*2056
    const int GU = 526336;

    dim3 blk(256);

    buf_update_kernel<<<960, blk, 0, stream>>>(inputs, tptr, objbuf, bufOut);
    zbin_kernel<<<960, blk, 0, stream>>>(bufOut, zb_in);

    // z_b = elu(zb_in @ Wb + bb) -> X[:, 0:1680]  (ldc = XD)
    gemm_wmma_f16<<<dim3(27, 4), blk, 0, stream>>>(zb_in, D_*F_, Wb, BD_, D_*F_, bb, X, XD_, 1);

    // first-layer projections (no bias, no act)
    gemm_wmma_f16<<<dim3(8, 4), blk, 0, stream>>>(cells, U_, Wr1a, HD_, U_,  nullptr, r_i, HD_, 0);
    gemm_wmma_f16<<<dim3(8, 4), blk, 0, stream>>>(cells, U_, Wr1b, HD_, U_,  nullptr, r_j, HD_, 0);
    gemm_wmma_f16<<<dim3(8, 4), blk, 0, stream>>>(cells, U_, Wl1a, HD_, U_,  nullptr, l_i, HD_, 0);
    gemm_wmma_f16<<<dim3(8, 4), blk, 0, stream>>>(X,    XD_, Wl1b, HD_, BD_, nullptr, l_j, HD_, 0);

    // fused pairwise MLP + reduction -> X[:, 1680:2704]
    pairwise_kernel<<<256, blk, 90112, stream>>>(r_i, r_j, l_i, l_j,
                                                 br1, Wr2, br2, Wr3, br3,
                                                 bl1, Wl2, bl2, Wl3, bl3, X);

    // LSTM gate pre-activations
    gemm_wmma_f16<<<dim3(33, 4), blk, 0, stream>>>(X, XD_, Wxi, U_, XD_, bxi, gx + 0*GU, U_, 0);
    gemm_wmma_f16<<<dim3(33, 4), blk, 0, stream>>>(X, XD_, Wxf, U_, XD_, bxf, gx + 1*GU, U_, 0);
    gemm_wmma_f16<<<dim3(33, 4), blk, 0, stream>>>(X, XD_, Wxo, U_, XD_, bxo, gx + 2*GU, U_, 0);
    gemm_wmma_f16<<<dim3(33, 4), blk, 0, stream>>>(X, XD_, Wxc, U_, XD_, bxc, gx + 3*GU, U_, 0);
    gemm_wmma_f16<<<dim3(33, 4), blk, 0, stream>>>(hidden, U_, Whi, U_, U_, bhi, gh + 0*GU, U_, 0);
    gemm_wmma_f16<<<dim3(33, 4), blk, 0, stream>>>(hidden, U_, Whf, U_, U_, bhf, gh + 1*GU, U_, 0);
    gemm_wmma_f16<<<dim3(33, 4), blk, 0, stream>>>(hidden, U_, Who, U_, U_, bho, gh + 2*GU, U_, 0);
    gemm_wmma_f16<<<dim3(33, 4), blk, 0, stream>>>(hidden, U_, Whc, U_, U_, bhc, gh + 3*GU, U_, 0);

    // elementwise LSTM combine -> H_new, C_new
    lstm_kernel<<<2056, blk, 0, stream>>>(gx + 0*GU, gx + 1*GU, gx + 2*GU, gx + 3*GU,
                                          gh + 0*GU, gh + 1*GU, gh + 2*GU, gh + 3*GU,
                                          cells, Hout, Cout);

    // output = H_new @ Wd + bd
    gemm_wmma_f16<<<dim3(1, 4), blk, 0, stream>>>(Hout, U_, Wd, D_, U_, bd, outDec, D_, 0);
}
